// NeuronBasedValue_10419590660470
// MI455X (gfx1250) — matrix operand order repaired and checked by reading, not compiled, not run on hardware
//
#include <hip/hip_runtime.h>
#include <math.h>

// ---------------- types ----------------
typedef __attribute__((ext_vector_type(16))) __bf16 v16bf;
typedef __attribute__((ext_vector_type(8)))  float  v8f;

union FragBF { uint4 u4[2]; v16bf v; };

// ---------------- problem dims (fixed by reference) ----------------
static constexpr int BATCH  = 2;
static constexpr int SSEQ   = 2048;
static constexpr int DDIM   = 1024;
static constexpr int HHEADS = 16;
static constexpr int NNEUR  = 256;
static constexpr int NBASIS = 64;
static constexpr int RRANK  = 128;
static constexpr int TOPK   = 8;
static constexpr int TTOK   = BATCH * SSEQ;      // 4096 tokens
static constexpr int VSIZE  = BATCH * HHEADS * SSEQ * 64;  // output V elements

__device__ __forceinline__ unsigned short f2bf(float f) {
    unsigned int u = __float_as_uint(f);
    unsigned int r = (u + 0x7FFFu + ((u >> 16) & 1u)) >> 16;
    return (unsigned short)r;
}

// ---------------- kernel 1: fp32 -> bf16 convert ----------------
__global__ void k_cvt_bf16(const float* __restrict__ src, unsigned short* __restrict__ dst, int n) {
    int i = blockIdx.x * blockDim.x + threadIdx.x;
    if (i < n) dst[i] = f2bf(src[i]);
}

// ---------------- kernel 2: basis_A [n,d,r] fp32 -> ABt [n,r,d] bf16 (LDS tiled) ----------------
__global__ void k_transpose_A(const float* __restrict__ A, unsigned short* __restrict__ ABt) {
    __shared__ float tile[32][33];
    int n  = blockIdx.z;
    int d0 = blockIdx.x * 32;
    int r0 = blockIdx.y * 32;
    const float* An = A + (size_t)n * DDIM * RRANK;
    #pragma unroll
    for (int i = 0; i < 4; ++i) {
        int d = threadIdx.y + 8 * i;
        tile[d][threadIdx.x] = An[(size_t)(d0 + d) * RRANK + r0 + threadIdx.x];
    }
    __syncthreads();
    unsigned short* Bn = ABt + (size_t)n * RRANK * DDIM;
    #pragma unroll
    for (int i = 0; i < 4; ++i) {
        int r = threadIdx.y + 8 * i;
        Bn[(size_t)(r0 + r) * DDIM + d0 + threadIdx.x] = f2bf(tile[threadIdx.x][r]);
    }
}

// ---------------- kernel 3: attention summary (mean over keys) ----------------
// attn [B,H,S,S] -> asum [T,H]  (t = b*S+s)
__global__ void k_attn_summary(const float* __restrict__ attn, float* __restrict__ asum) {
    int tid  = threadIdx.x;
    int wid  = (blockIdx.x * blockDim.x + tid) >> 5;   // one wave per (b,h,s): 0..65535
    int lane = tid & 31;
    const float4* row = (const float4*)(attn + (size_t)wid * SSEQ);
    float s = 0.f;
    #pragma unroll
    for (int i = 0; i < 16; ++i) {
        float4 v = row[lane + i * 32];
        s += v.x + v.y + v.z + v.w;
    }
    #pragma unroll
    for (int off = 16; off > 0; off >>= 1) s += __shfl_xor(s, off, 32);
    if (lane == 0) {
        int b = wid >> 15;
        int h = (wid >> 11) & 15;
        int sq = wid & 2047;
        asum[(size_t)(b * SSEQ + sq) * HHEADS + h] = s * (1.0f / SSEQ);
    }
}

// ---------------- kernel 4: recipe softmax + neuron_emb = softmax(recipe) @ basis_emb ----------------
__global__ void k_neuron_prep(const float* __restrict__ recipe, const float* __restrict__ basis_emb,
                              float* __restrict__ recipe_norm, unsigned short* __restrict__ neb) {
    __shared__ float red[64];
    __shared__ float rn[64];
    int n = blockIdx.x;
    int t = threadIdx.x;                      // 64 threads
    float v = recipe[n * NBASIS + t];
    red[t] = v; __syncthreads();
    for (int off = 32; off > 0; off >>= 1) { if (t < off) red[t] = fmaxf(red[t], red[t + off]); __syncthreads(); }
    float m = red[0]; __syncthreads();
    float e = __expf(v - m);
    red[t] = e; __syncthreads();
    for (int off = 32; off > 0; off >>= 1) { if (t < off) red[t] += red[t + off]; __syncthreads(); }
    float r = e / red[0];
    rn[t] = r;
    recipe_norm[n * NBASIS + t] = r;
    __syncthreads();
    #pragma unroll 4
    for (int i = 0; i < 16; ++i) {
        int d = t + 64 * i;
        float acc = 0.f;
        for (int j = 0; j < 64; ++j) acc += rn[j] * basis_emb[j * DDIM + d];
        neb[(size_t)n * DDIM + d] = f2bf(acc);
    }
}

// ---------------- kernel 5: semantic WMMA GEMM + context gate + top-8 + token_recipe ----------------
// per block: 16 tokens x 256 neurons
__global__ void k_route(const unsigned short* __restrict__ xb,
                        const unsigned short* __restrict__ neb,
                        const float* __restrict__ asum,
                        const float* __restrict__ ncp,
                        const float* __restrict__ recipe_norm,
                        float* __restrict__ token_recipe,
                        int* __restrict__ idx_out) {
    __shared__ float asum_s[16][16];
    __shared__ float ncp_s[NNEUR][16];
    __shared__ float fs[16][NNEUR + 1];
    __shared__ float wv_s[16][TOPK];
    __shared__ int   idx_s[16][TOPK];

    int tid = threadIdx.x;
    int t0  = blockIdx.x * 16;

    asum_s[tid >> 4][tid & 15] = asum[(size_t)(t0 + (tid >> 4)) * HHEADS + (tid & 15)];
    #pragma unroll
    for (int i = 0; i < 16; ++i) {
        int id = tid + 256 * i;
        ncp_s[id >> 4][id & 15] = ncp[id];
    }
    __syncthreads();

    int w = tid >> 5, lane = tid & 31;
    int lm = lane & 15, hi = lane >> 4;
    int bkA = hi ? 16 : 0;    // A frag K split
    int kbB = hi ? 16 : 0;    // B frag K base
    int tb  = hi ? 8 : 0;     // C frag M base
    int nt0 = w * 32;         // each wave: two 16-neuron tiles

    v8f c0 = {}; v8f c1 = {};
    const unsigned short* paRow  = xb  + (size_t)(t0 + lm) * DDIM;
    const unsigned short* pb0Row = neb + (size_t)(nt0 + lm) * DDIM;
    const unsigned short* pb1Row = neb + (size_t)(nt0 + 16 + lm) * DDIM;

    for (int kb = 0; kb < DDIM / 32; ++kb) {
        int d0 = kb * 32;
        FragBF a, b0, b1;
        const unsigned short* pa = paRow + d0 + bkA;
        a.u4[0] = *(const uint4*)pa;
        a.u4[1] = *(const uint4*)(pa + 8);
        const unsigned short* p0 = pb0Row + d0 + kbB;
        b0.u4[0] = *(const uint4*)p0;
        b0.u4[1] = *(const uint4*)(p0 + 8);
        const unsigned short* p1 = pb1Row + d0 + kbB;
        b1.u4[0] = *(const uint4*)p1;
        b1.u4[1] = *(const uint4*)(p1 + 8);
        c0 = __builtin_amdgcn_wmma_f32_16x16x32_bf16(false, a.v, false, b0.v, (short)0, c0, false, false);
        c1 = __builtin_amdgcn_wmma_f32_16x16x32_bf16(false, a.v, false, b1.v, (short)0, c1, false, false);
    }

    // context gate + write final scores
    #pragma unroll
    for (int v = 0; v < 8; ++v) {
        int tl = tb + v;
        {
            int n = nt0 + lm;
            float ctx = 0.f;
            #pragma unroll
            for (int h = 0; h < 16; ++h) ctx += asum_s[tl][h] * ncp_s[n][h];
            fs[tl][n] = c0[v] * (1.f / (1.f + __expf(-ctx)));
        }
        {
            int n = nt0 + 16 + lm;
            float ctx = 0.f;
            #pragma unroll
            for (int h = 0; h < 16; ++h) ctx += asum_s[tl][h] * ncp_s[n][h];
            fs[tl][n] = c1[v] * (1.f / (1.f + __expf(-ctx)));
        }
    }
    __syncthreads();

    // top-8 per token (threads 0..15, one token each)
    if (tid < 16) {
        float vals[TOPK]; int idxs[TOPK];
        for (int k = 0; k < TOPK; ++k) {
            float best = -1e30f; int bi = 0;
            for (int j = 0; j < NNEUR; ++j) {
                float f = fs[tid][j];
                if (f > best) { best = f; bi = j; }
            }
            vals[k] = best; idxs[k] = bi;
            fs[tid][bi] = -1e30f;
        }
        float m = vals[0], se = 0.f, ex[TOPK];
        #pragma unroll
        for (int k = 0; k < TOPK; ++k) { ex[k] = __expf(vals[k] - m); se += ex[k]; }
        float inv = 1.f / se;
        #pragma unroll
        for (int k = 0; k < TOPK; ++k) {
            wv_s[tid][k]  = ex[k] * inv;
            idx_s[tid][k] = idxs[k];
            idx_out[(t0 + tid) * TOPK + k] = idxs[k];
        }
    }
    __syncthreads();

    // token_recipe[t, nb] = sum_k w_k * recipe_norm[idx_k, nb]
    #pragma unroll
    for (int i = 0; i < 4; ++i) {
        int id = tid + 256 * i;          // 0..1023 = 16 tokens x 64 bases
        int tl = id >> 6, nb = id & 63;
        float acc = 0.f;
        #pragma unroll
        for (int k = 0; k < TOPK; ++k)
            acc += wv_s[tl][k] * recipe_norm[idx_s[tl][k] * NBASIS + nb];
        token_recipe[(size_t)(t0 + tl) * NBASIS + nb] = acc;
    }
}

// ---------------- kernel 6: big GEMM  v_sem[t,r] = sum_n c[t,n] * (X @ A_n)[t,r] ----------------
// block: 32 tokens x 128 r; 8 waves, wave = (m_sub in {0,1}) x (2 r-tiles of 16)
static constexpr int MT  = 32;
static constexpr int DCH = 512;
__global__ void k_big_gemm(const unsigned short* __restrict__ xb,
                           const unsigned short* __restrict__ ABt,
                           const float* __restrict__ tr,
                           unsigned short* __restrict__ vsem) {
    __shared__ unsigned short xs[MT][DCH + 8];   // padded rows: stride 1040B, 16B aligned, bank-spread
    __shared__ float cs[MT][NBASIS];
    int tid = threadIdx.x;
    int t0  = blockIdx.x * MT;

    #pragma unroll
    for (int i = 0; i < (MT * NBASIS) / 256; ++i) {          // 8
        int id = tid + 256 * i;
        cs[id >> 6][id & 63] = tr[(size_t)(t0 + (id >> 6)) * NBASIS + (id & 63)];
    }

    int w = tid >> 5, lane = tid & 31;
    int lm = lane & 15, hi = lane >> 4;
    int m0 = (w >> 2) * 16;
    int r0 = (w & 3) * 32;
    int bkA = hi ? 16 : 0, kbB = hi ? 16 : 0, tb = hi ? 8 : 0;

    v8f vs0 = {}; v8f vs1 = {};

    for (int ch = 0; ch < DDIM / DCH; ++ch) {
        __syncthreads();
        #pragma unroll
        for (int i = 0; i < (MT * DCH / 8) / 256; ++i) {     // 8 x uint4 per thread
            int id  = tid + 256 * i;
            int row = id >> 6;                               // DCH/8 = 64 per row
            int c8  = id & 63;
            *(uint4*)&xs[row][c8 * 8] =
                *(const uint4*)(xb + (size_t)(t0 + row) * DDIM + ch * DCH + c8 * 8);
        }
        __syncthreads();

        for (int n = 0; n < NBASIS; ++n) {
            v8f p0 = {}; v8f p1 = {};
            const unsigned short* Bn = ABt + (size_t)n * RRANK * DDIM + ch * DCH;
            const unsigned short* b0row = Bn + (size_t)(r0 + lm) * DDIM;
            const unsigned short* b1row = Bn + (size_t)(r0 + 16 + lm) * DDIM;
            for (int kb = 0; kb < DCH / 32; ++kb) {
                int d0 = kb * 32;
                FragBF a, b0, b1;
                const unsigned short* pa = &xs[m0 + lm][d0 + bkA];
                a.u4[0] = *(const uint4*)pa;
                a.u4[1] = *(const uint4*)(pa + 8);
                const unsigned short* pb0 = b0row + d0 + kbB;
                b0.u4[0] = *(const uint4*)pb0;
                b0.u4[1] = *(const uint4*)(pb0 + 8);
                const unsigned short* pb1 = b1row + d0 + kbB;
                b1.u4[0] = *(const uint4*)pb1;
                b1.u4[1] = *(const uint4*)(pb1 + 8);
                p0 = __builtin_amdgcn_wmma_f32_16x16x32_bf16(false, a.v, false, b0.v, (short)0, p0, false, false);
                p1 = __builtin_amdgcn_wmma_f32_16x16x32_bf16(false, a.v, false, b1.v, (short)0, p1, false, false);
            }
            #pragma unroll
            for (int v = 0; v < 8; ++v) {
                float sc = cs[m0 + tb + v][n];
                vs0[v] += sc * p0[v];
                vs1[v] += sc * p1[v];
            }
        }
    }

    #pragma unroll
    for (int v = 0; v < 8; ++v) {
        int t = t0 + m0 + tb + v;
        vsem[(size_t)t * RRANK + r0 + lm]      = f2bf(vs0[v]);
        vsem[(size_t)t * RRANK + r0 + 16 + lm] = f2bf(vs1[v]);
    }
}

// ---------------- kernel 7: out GEMM  V = v_sem @ v_out_w^T + b, permuted store ----------------
__global__ void k_out_gemm(const unsigned short* __restrict__ vsem,
                           const unsigned short* __restrict__ vwb,
                           const float* __restrict__ bias,
                           float* __restrict__ out) {
    int tid = threadIdx.x;
    int w = tid >> 5, lane = tid & 31;
    int lm = lane & 15, hi = lane >> 4;
    int m0 = blockIdx.x * 16;
    int dout0 = blockIdx.y * 128 + w * 16;
    int bkA = hi ? 16 : 0, kbB = hi ? 16 : 0, tb = hi ? 8 : 0;
    v8f acc = {};
    #pragma unroll
    for (int kb = 0; kb < RRANK / 32; ++kb) {    // 4
        int d0 = kb * 32;
        FragBF a, b;
        const unsigned short* pa = vsem + (size_t)(m0 + lm) * RRANK + d0 + bkA;
        a.u4[0] = *(const uint4*)pa;
        a.u4[1] = *(const uint4*)(pa + 8);
        const unsigned short* pb = vwb + (size_t)(dout0 + lm) * RRANK + d0 + kbB;
        b.u4[0] = *(const uint4*)pb;
        b.u4[1] = *(const uint4*)(pb + 8);
        acc = __builtin_amdgcn_wmma_f32_16x16x32_bf16(false, a.v, false, b.v, (short)0, acc, false, false);
    }
    int dout = dout0 + lm;
    float bv = bias[dout];
    int h = dout >> 6, dd = dout & 63;
    #pragma unroll
    for (int v = 0; v < 8; ++v) {
        int t  = m0 + tb + v;
        int bb = t >> 11, s = t & 2047;
        out[(((size_t)bb * HHEADS + h) * SSEQ + s) * 64 + dd] = acc[v] + bv;
    }
}

// ---------------- launcher ----------------
extern "C" void kernel_launch(void* const* d_in, const int* in_sizes, int n_in,
                              void* d_out, int out_size, void* d_ws, size_t ws_size,
                              hipStream_t stream) {
    const float* x         = (const float*)d_in[0];
    const float* attn      = (const float*)d_in[1];
    const float* recipe    = (const float*)d_in[2];
    const float* ncp       = (const float*)d_in[3];
    const float* basis_emb = (const float*)d_in[4];
    const float* basis_A   = (const float*)d_in[5];
    const float* v_out_w   = (const float*)d_in[6];
    const float* v_out_b   = (const float*)d_in[7];
    float* out = (float*)d_out;
    int* idx_out = (int*)d_out + (size_t)VSIZE;

    char* p = (char*)d_ws;
    auto alloc = [&](size_t bytes) -> char* {
        char* r = p;
        p += (bytes + 255) & ~(size_t)255;
        return r;
    };
    unsigned short* xb   = (unsigned short*)alloc(sizeof(unsigned short) * TTOK * DDIM);      // 8 MB
    unsigned short* ABt  = (unsigned short*)alloc(sizeof(unsigned short) * NBASIS * RRANK * DDIM); // 16 MB
    unsigned short* neb  = (unsigned short*)alloc(sizeof(unsigned short) * NNEUR * DDIM);     // 512 KB
    unsigned short* vwb  = (unsigned short*)alloc(sizeof(unsigned short) * DDIM * RRANK);     // 256 KB
    float* recipe_norm   = (float*)alloc(sizeof(float) * NNEUR * NBASIS);                     // 64 KB
    float* asum          = (float*)alloc(sizeof(float) * TTOK * HHEADS);                      // 256 KB
    float* token_recipe  = (float*)alloc(sizeof(float) * TTOK * NBASIS);                      // 1 MB
    unsigned short* vsem = (unsigned short*)alloc(sizeof(unsigned short) * TTOK * RRANK);     // 1 MB

    int nX = TTOK * DDIM;        // 4194304
    int nW = DDIM * RRANK;       // 131072
    k_cvt_bf16<<<(nX + 255) / 256, 256, 0, stream>>>(x, xb, nX);
    k_cvt_bf16<<<(nW + 255) / 256, 256, 0, stream>>>(v_out_w, vwb, nW);
    k_transpose_A<<<dim3(DDIM / 32, RRANK / 32, NBASIS), dim3(32, 8), 0, stream>>>(basis_A, ABt);
    k_attn_summary<<<(BATCH * HHEADS * SSEQ) / 8, 256, 0, stream>>>(attn, asum);
    k_neuron_prep<<<NNEUR, 64, 0, stream>>>(recipe, basis_emb, recipe_norm, neb);
    k_route<<<TTOK / 16, 256, 0, stream>>>(xb, neb, asum, ncp, recipe_norm, token_recipe, idx_out);
    k_big_gemm<<<TTOK / MT, 256, 0, stream>>>(xb, ABt, token_recipe, vsem);
    k_out_gemm<<<dim3(TTOK / 16, DDIM / 128), 256, 0, stream>>>(vsem, vwb, v_out_b, out);
}
